// self_att_layer_68573447848478
// MI455X (gfx1250) — matrix operand order repaired and checked by reading, not compiled, not run on hardware
//
#include <hip/hip_runtime.h>
#include <hip/hip_bf16.h>
#include <math.h>

#define NROWS 8192
#define EMB2  512
#define HID   128
#define JT    32      // K tile
#define RT    32      // rows per workgroup (2 M-tiles)
#define LOG2E 1.44269504088896340736f

typedef __attribute__((ext_vector_type(16))) __bf16        v16bf;
typedef __attribute__((ext_vector_type(8)))  float         v8f;
typedef __attribute__((ext_vector_type(4)))  unsigned int  u32x4;

union V16 { v16bf bf; u32x4 u[2]; };

static __device__ __forceinline__ unsigned short f32_to_bf16(float x) {
    union { float f; unsigned int u; } v; v.f = x;
    unsigned int r = v.u + 0x7FFFu + ((v.u >> 16) & 1u);  // round-to-nearest-even
    return (unsigned short)(r >> 16);
}

// ---- prep 1: u2 = W1^T w2, u3 = W1^T w3 ----------------------------------
__global__ void k_proj_vec(const float* __restrict__ W1, const float* __restrict__ w2,
                           const float* __restrict__ w3, float* __restrict__ u2,
                           float* __restrict__ u3) {
    int e = blockIdx.x * blockDim.x + threadIdx.x;
    if (e >= EMB2) return;
    float a = 0.f, b = 0.f;
    for (int h = 0; h < HID; ++h) {
        float w = W1[h * EMB2 + e];
        a += w * w2[h];
        b += w * w3[h];
    }
    u2[e] = a; u3[e] = b;
}

// ---- prep 2: f1[n] = in[n]·u2 + b2 ; f2[n] = in[n]·u3 + b3 ---------------
__global__ void k_scalars(const float* __restrict__ in, const float* __restrict__ u2,
                          const float* __restrict__ u3, const float* __restrict__ b2,
                          const float* __restrict__ b3, float* __restrict__ f1,
                          float* __restrict__ f2) {
    int n = blockIdx.x * blockDim.x + threadIdx.x;
    const float4* row = (const float4*)(in + (size_t)n * EMB2);
    const float4* U2  = (const float4*)u2;
    const float4* U3  = (const float4*)u3;
    float a = 0.f, b = 0.f;
    for (int e = 0; e < EMB2 / 4; ++e) {
        float4 x = row[e], p = U2[e], q = U3[e];
        a += x.x * p.x + x.y * p.y + x.z * p.z + x.w * p.w;
        b += x.x * q.x + x.y * q.y + x.z * q.z + x.w * q.w;
    }
    f1[n] = a + b2[0];
    f2[n] = b + b3[0];
}

// ---- prep 3: Vt[c][j] = bf16(in[j][c])  (tiled transpose) ----------------
__global__ void k_transpose_bf16(const float* __restrict__ in, unsigned short* __restrict__ vt) {
    __shared__ float tile[32][33];
    int j0 = blockIdx.x * 32, c0 = blockIdx.y * 32;
    int tx = threadIdx.x, ty = threadIdx.y;
    #pragma unroll
    for (int k = 0; k < 4; ++k)
        tile[ty + 8 * k][tx] = in[(size_t)(j0 + ty + 8 * k) * EMB2 + c0 + tx];
    __syncthreads();
    #pragma unroll
    for (int k = 0; k < 4; ++k)
        vt[(size_t)(c0 + ty + 8 * k) * NROWS + j0 + tx] = f32_to_bf16(tile[tx][ty + 8 * k]);
}

// ---- main: fused masked-softmax attention, P·V via bf16 WMMA -------------
__global__ __launch_bounds__(256)
void k_attn(const int* __restrict__ adj, const float* __restrict__ f1,
            const float* __restrict__ f2, const unsigned short* __restrict__ vt,
            float* __restrict__ out) {
    __shared__ unsigned int Pb[2][RT * JT / 2];  // bf16 P tile, double buffered
    __shared__ float        Sb[2][RT];           // per-row rescale factors
    __shared__ float        Lb[RT];              // final softmax sums

    const int tid = threadIdx.x;
    const int w   = tid >> 5;        // wave 0..7 -> owns columns 64w..64w+63
    const int l   = tid & 31;
    const int h2  = l >> 4;          // lane half
    const int n   = l & 15;          // N index within a 16-col tile
    const int rowbase = blockIdx.x * RT;

    // producer role: wave w produces score rows 4w..4w+3; 8 lanes per row
    const int   kq   = l & 7;              // K sub-quad: covers k = 4kq..4kq+3
    const int   rloc = 4 * w + (l >> 3);   // local row 0..31
    const int   grow = rowbase + rloc;
    const float f1r  = f1[grow];
    const int*  adjrow = adj + (size_t)grow * NROWS;

    float m = -1e30f, lsum = 0.f;

    v8f acc[2][4];
    #pragma unroll
    for (int mt = 0; mt < 2; ++mt)
        #pragma unroll
        for (int g = 0; g < 4; ++g)
            #pragma unroll
            for (int e = 0; e < 8; ++e) acc[mt][g][e] = 0.f;

    const unsigned short* vbase[4];
    #pragma unroll
    for (int g = 0; g < 4; ++g)
        vbase[g] = vt + (size_t)(64 * w + 16 * g + n) * NROWS;

    for (int t = 0; t < NROWS / JT; ++t) {
        const int j0  = t * JT;
        const int buf = t & 1;

        // ---------------- produce P tile (online softmax) ----------------
        u32x4  a4 = *(const u32x4*)(adjrow + j0 + 4 * kq);
        float4 fj = *(const float4*)(f2 + j0 + 4 * kq);
        float s0, s1, s2, s3;
        {
            float x0 = f1r + fj.x, x1 = f1r + fj.y, x2 = f1r + fj.z, x3 = f1r + fj.w;
            float r0 = fmaxf(x0, 0.01f * x0), r1 = fmaxf(x1, 0.01f * x1);
            float r2 = fmaxf(x2, 0.01f * x2), r3 = fmaxf(x3, 0.01f * x3);
            s0 = (a4[0] != 0u) ? r0 : -1e9f;
            s1 = (a4[1] != 0u) ? r1 : -1e9f;
            s2 = (a4[2] != 0u) ? r2 : -1e9f;
            s3 = (a4[3] != 0u) ? r3 : -1e9f;
        }
        float tmax = fmaxf(fmaxf(s0, s1), fmaxf(s2, s3));
        tmax = fmaxf(tmax, __shfl_xor(tmax, 1, 32));
        tmax = fmaxf(tmax, __shfl_xor(tmax, 2, 32));
        tmax = fmaxf(tmax, __shfl_xor(tmax, 4, 32));
        const float mnew  = fmaxf(m, tmax);
        const float scale = exp2f((m - mnew) * LOG2E);
        const float p0 = exp2f((s0 - mnew) * LOG2E);
        const float p1 = exp2f((s1 - mnew) * LOG2E);
        const float p2 = exp2f((s2 - mnew) * LOG2E);
        const float p3 = exp2f((s3 - mnew) * LOG2E);
        float ts = p0 + p1 + p2 + p3;
        ts += __shfl_xor(ts, 1, 32);
        ts += __shfl_xor(ts, 2, 32);
        ts += __shfl_xor(ts, 4, 32);
        lsum = lsum * scale + ts;
        m = mnew;
        Pb[buf][rloc * 16 + 2 * kq]     = (unsigned int)f32_to_bf16(p0) | ((unsigned int)f32_to_bf16(p1) << 16);
        Pb[buf][rloc * 16 + 2 * kq + 1] = (unsigned int)f32_to_bf16(p2) | ((unsigned int)f32_to_bf16(p3) << 16);
        if (kq == 0) Sb[buf][rloc] = scale;

        __syncthreads();

        // ---------------- consume: rescale, A/B fragments, WMMA ----------
        #pragma unroll
        for (int mt = 0; mt < 2; ++mt) {
            const float* sp = &Sb[buf][16 * mt + 8 * h2];   // C layout: M = e + 8*h2
            float sc[8];
            *(float4*)&sc[0] = *(const float4*)sp;
            *(float4*)&sc[4] = *(const float4*)(sp + 4);
            #pragma unroll
            for (int g = 0; g < 4; ++g)
                #pragma unroll
                for (int e = 0; e < 8; ++e) acc[mt][g][e] *= sc[e];
        }

        V16 A[2];
        #pragma unroll
        for (int mt = 0; mt < 2; ++mt) {
            // A layout: lane row = l&15 (+16*mt); VGPRs 0-3 hold K = 8*h2..8*h2+7,
            // VGPRs 4-7 hold K = 16+8*h2..16+8*h2+7 (packed pairs, K contiguous)
            const unsigned int* pp = &Pb[buf][(16 * mt + (l & 15)) * 16 + 4 * h2];
            A[mt].u[0] = *(const u32x4*)pp;
            A[mt].u[1] = *(const u32x4*)(pp + 8);
        }

        #pragma unroll
        for (int g = 0; g < 4; ++g) {
            // B layout: lane col = l&15; elems 0-7 hold K 8*h2..8*h2+7, elems 8-15 K 16+8*h2..16+8*h2+7
            V16 B;
            const unsigned short* vp = vbase[g] + j0 + 8 * h2;
            B.u[0] = *(const u32x4*)vp;
            B.u[1] = *(const u32x4*)(vp + 16);
            acc[0][g] = __builtin_amdgcn_wmma_f32_16x16x32_bf16(
                false, A[0].bf, false, B.bf, (short)0, acc[0][g], false, false);
            acc[1][g] = __builtin_amdgcn_wmma_f32_16x16x32_bf16(
                false, A[1].bf, false, B.bf, (short)0, acc[1][g], false, false);
        }
    }

    // ---------------- epilogue: divide by softmax sum, ReLU, store -------
    if (kq == 0) Lb[rloc] = lsum;
    __syncthreads();

    #pragma unroll
    for (int mt = 0; mt < 2; ++mt) {
        float li[8], inv[8];
        *(float4*)&li[0] = *(const float4*)&Lb[16 * mt + 8 * h2];
        *(float4*)&li[4] = *(const float4*)&Lb[16 * mt + 8 * h2 + 4];
        #pragma unroll
        for (int e = 0; e < 8; ++e) inv[e] = 1.0f / li[e];
        #pragma unroll
        for (int g = 0; g < 4; ++g) {
            const int c = 64 * w + 16 * g + n;
            #pragma unroll
            for (int e = 0; e < 8; ++e) {
                const int row = rowbase + 16 * mt + 8 * h2 + e;
                float val = acc[mt][g][e] * inv[e];
                out[(size_t)row * EMB2 + c] = fmaxf(val, 0.f);
            }
        }
    }
}

extern "C" void kernel_launch(void* const* d_in, const int* in_sizes, int n_in,
                              void* d_out, int out_size, void* d_ws, size_t ws_size,
                              hipStream_t stream) {
    (void)in_sizes; (void)n_in; (void)out_size; (void)ws_size;
    const float* inlayer = (const float*)d_in[0];
    const int*   adj     = (const int*)d_in[1];
    const float* W1      = (const float*)d_in[2];
    const float* w2      = (const float*)d_in[3];
    const float* b2      = (const float*)d_in[4];
    const float* w3      = (const float*)d_in[5];
    const float* b3      = (const float*)d_in[6];
    float*       out     = (float*)d_out;

    // workspace layout: f1(8192f) f2(8192f) u2(512f) u3(512f) Vt(512*8192 bf16)
    float*          f1 = (float*)d_ws;
    float*          f2 = f1 + NROWS;
    float*          u2 = f2 + NROWS;
    float*          u3 = u2 + EMB2;
    unsigned short* vt = (unsigned short*)(u3 + EMB2);

    k_proj_vec<<<(EMB2 + 255) / 256, 256, 0, stream>>>(W1, w2, w3, u2, u3);
    k_scalars<<<NROWS / 256, 256, 0, stream>>>(inlayer, u2, u3, b2, b3, f1, f2);
    dim3 tb(32, 8);
    dim3 tg(NROWS / 32, EMB2 / 32);
    k_transpose_bf16<<<tg, tb, 0, stream>>>(inlayer, vt);
    k_attn<<<NROWS / RT, 256, 0, stream>>>(adj, f1, f2, vt, out);
}
